// VectorNetBackbone_44006234914856
// MI455X (gfx1250) — matrix operand order repaired and checked by reading, not compiled, not run on hardware
//
#include <hip/hip_runtime.h>

// ---------------------------------------------------------------------------
// VectorNet backbone on gfx1250: all matmuls via v_wmma_f32_16x16x32_bf16.
// fp32 inputs are converted to bf16 for A/B, accumulated in f32.
// GEMM is templated on K-chunks / LDS usage / epilogue so the inner loop is a
// branch-free, fully unrolled WMMA chain.
// ---------------------------------------------------------------------------

typedef __attribute__((ext_vector_type(16))) __bf16 bf16x16;
typedef __attribute__((ext_vector_type(8)))  float  floatx8;

union Frag { bf16x16 v; uint4 q[2]; };

// ---------------------------------------------------------------------------
// Pack a (Kreal x 128) f32 weight into WMMA B-fragment layout, bf16, K padded
// to Kc*32.  Layout: [kc][nt][lane][16 elems]; lane l holds element j at
// k = kc*32 + 16*(l/16) + j, n = nt*16 + (l%16).
// ---------------------------------------------------------------------------
__global__ void vn_pack_weight(const float* __restrict__ W, __bf16* __restrict__ out,
                               int Kreal, int Kc) {
    int idx = blockIdx.x * blockDim.x + threadIdx.x;
    int total = Kc * 8 * 32;
    if (idx >= total) return;
    int lane = idx & 31;
    int nt   = (idx >> 5) & 7;
    int kc   = idx >> 8;
    int n     = nt * 16 + (lane & 15);
    int kbase = kc * 32 + 16 * (lane >> 4);
    __bf16* o = out + (size_t)idx * 16;
#pragma unroll
    for (int j = 0; j < 16; ++j) {
        int k = kbase + j;
        o[j] = (k < Kreal) ? (__bf16)W[(size_t)k * 128 + n] : (__bf16)0.0f;
    }
}

// ---------------------------------------------------------------------------
// Center points on last point's xy, convert to bf16, pad K: 8 -> 32.
// ---------------------------------------------------------------------------
__global__ void vn_center_pad(const float* __restrict__ pts, __bf16* __restrict__ xc,
                              int Nn, int P) {
    int row = blockIdx.x * blockDim.x + threadIdx.x;   // node*P + p
    if (row >= Nn * P) return;
    int n = row / P;
    const float* x   = pts + (size_t)row * 8;
    const float* ref = pts + ((size_t)n * P + (P - 1)) * 8;
    __bf16* o = xc + (size_t)row * 32;
    o[0] = (__bf16)(x[0] - ref[0]);
    o[1] = (__bf16)(x[1] - ref[1]);
#pragma unroll
    for (int i = 2; i < 8; ++i) o[i] = (__bf16)x[i];
#pragma unroll
    for (int i = 8; i < 32; ++i) o[i] = (__bf16)0.0f;
}

// ---------------------------------------------------------------------------
// WMMA GEMM: out = relu(A[M x KC*32] @ Wpacked + bias) (+ baseF32 residual).
// A: bf16 row-major, leading dim lda (elements).  Output leading dim = 128.
// Block = 256 threads (8 waves); each wave computes a 16x128 strip.
// KC, LDS usage and epilogue behavior are compile-time -> branch-free inner
// loop of KC*8 back-to-back v_wmma_f32_16x16x32_bf16.
// ---------------------------------------------------------------------------
template <int KC, bool USE_LDS, bool OUT_F, bool OUT_B, bool RES>
__launch_bounds__(256)
__global__ void vn_gemm(const __bf16* __restrict__ A, int lda,
                        const __bf16* __restrict__ Wp,
                        const float* __restrict__ bias, int M,
                        float* __restrict__ outF32,
                        __bf16* __restrict__ outBF16,
                        const float* __restrict__ baseF32) {
    __shared__ uint4 sW[USE_LDS ? KC * 512 : 1];   // packed B frags (8 KB / kc)
    __shared__ float sBias[128];

    const int tid = threadIdx.x;
    if constexpr (USE_LDS) {
        const uint4* g = (const uint4*)Wp;
#pragma unroll
        for (int i = 0; i < KC * 2; ++i) sW[i * 256 + tid] = g[i * 256 + tid];
    }
    if (tid < 128) sBias[tid] = bias[tid];
    __syncthreads();

    const int wave = tid >> 5;
    const int lane = tid & 31;
    const int half = lane >> 4;
    const int m0   = blockIdx.x * 128 + wave * 16;
    const int rowL = m0 + (lane & 15);
    const int row  = (rowL < M) ? rowL : (M - 1);   // clamp, discard at store

    // Prefetch next block's slice of the streamed A matrix (global_prefetch_b8).
    if (rowL + 128 < M) __builtin_prefetch(A + (size_t)(rowL + 128) * lda, 0, 3);

    floatx8 acc[8];
#pragma unroll
    for (int nt = 0; nt < 8; ++nt) acc[nt] = (floatx8){0, 0, 0, 0, 0, 0, 0, 0};

    const uint4* gW = (const uint4*)Wp;
#pragma unroll
    for (int kc = 0; kc < KC; ++kc) {
        // A fragment (16x32 bf16): lane half selects contiguous K half (matches packed B)
        Frag a;
        const __bf16* ap = A + (size_t)row * lda + kc * 32 + half * 16;
        a.q[0] = *(const uint4*)(ap);        // K = base+0..7   (elems 0..7)
        a.q[1] = *(const uint4*)(ap + 8);    // K = base+8..15  (elems 8..15)
#pragma unroll
        for (int nt = 0; nt < 8; ++nt) {
            Frag b;
            const int fi = ((kc * 8 + nt) * 32 + lane) * 2;
            if constexpr (USE_LDS) { b.q[0] = sW[fi]; b.q[1] = sW[fi + 1]; }
            else                   { b.q[0] = gW[fi]; b.q[1] = gW[fi + 1]; }
            acc[nt] = __builtin_amdgcn_wmma_f32_16x16x32_bf16(
                false, a.v, false, b.v, (short)0, acc[nt], false, false);
        }
    }

    // Epilogue: bias + relu (+ residual), dual f32/bf16 stores.
    const int nb    = lane & 15;
    const int mBase = m0 + 8 * half;       // C/D layout: VGPR r -> M = r + 8*(lane/16)
    auto emit = [&](int nt, int r) {
        const int n = nt * 16 + nb;
        float v = acc[nt][r] + sBias[n];
        v = v > 0.0f ? v : 0.0f;
        const size_t oi = (size_t)(mBase + r) * 128 + n;
        if constexpr (RES)   v += baseF32[oi];
        if constexpr (OUT_F) outF32[oi] = v;
        if constexpr (OUT_B) outBF16[oi] = (__bf16)v;
    };
    if (m0 + 16 <= M) {                    // full tile: unguarded stores
#pragma unroll
        for (int nt = 0; nt < 8; ++nt)
#pragma unroll
            for (int r = 0; r < 8; ++r) emit(nt, r);
    } else {                               // edge tile
#pragma unroll
        for (int nt = 0; nt < 8; ++nt)
#pragma unroll
            for (int r = 0; r < 8; ++r)
                if (mBase + r < M) emit(nt, r);
    }
}

// ---------------------------------------------------------------------------
// Max-pool over P points per node (bf16 in / bf16 out).
// ---------------------------------------------------------------------------
__global__ void vn_maxpool(const __bf16* __restrict__ h, __bf16* __restrict__ out,
                           int Nn, int P) {
    int i = blockIdx.x * blockDim.x + threadIdx.x;   // n*128 + c
    if (i >= Nn * 128) return;
    int n = i >> 7, c = i & 127;
    const __bf16* p = h + ((size_t)n * P) * 128 + c;
    float m = -3.0e38f;
    for (int t = 0; t < P; ++t) {
        float v = (float)p[(size_t)t * 128];
        m = v > m ? v : m;
    }
    out[i] = (__bf16)m;
}

// ---------------------------------------------------------------------------
// Edge scatter: agg[dst] += feat[src]; deg[dst] += 1.  f32 global atomics.
// ---------------------------------------------------------------------------
__global__ void vn_scatter_add(const int* __restrict__ src, const int* __restrict__ dst,
                               int E, const __bf16* __restrict__ feat,
                               float* __restrict__ agg, float* __restrict__ deg) {
    long t = (long)blockIdx.x * blockDim.x + threadIdx.x;
    if (t >= (long)E * 128) return;
    int e = (int)(t >> 7), c = (int)(t & 127);
    int s = src[e], d = dst[e];
    atomicAdd(&agg[(size_t)d * 128 + c], (float)feat[(size_t)s * 128 + c]);
    if (c == 0) atomicAdd(&deg[d], 1.0f);
}

// ---------------------------------------------------------------------------
// concat([node, agg/max(deg,1)]) in bf16, row length 256.
// ---------------------------------------------------------------------------
__global__ void vn_concat(const __bf16* __restrict__ node, const float* __restrict__ agg,
                          const float* __restrict__ deg, __bf16* __restrict__ out, int Nn) {
    int i = blockIdx.x * blockDim.x + threadIdx.x;
    if (i >= Nn * 128) return;
    int n = i >> 7, c = i & 127;
    float dg = deg[n]; dg = dg > 1.0f ? dg : 1.0f;
    out[(size_t)n * 256 + c]       = node[i];
    out[(size_t)n * 256 + 128 + c] = (__bf16)(agg[i] / dg);
}

__global__ void vn_zero_f32(float* __restrict__ p, long n) {
    long i = (long)blockIdx.x * blockDim.x + threadIdx.x;
    if (i < n) p[i] = 0.0f;
}

// ---------------------------------------------------------------------------
// Host orchestration
// ---------------------------------------------------------------------------
static inline int cdiv(long a, long b) { return (int)((a + b - 1) / b); }

extern "C" void kernel_launch(void* const* d_in, const int* in_sizes, int n_in,
                              void* d_out, int out_size, void* d_ws, size_t ws_size,
                              hipStream_t stream) {
    const int P = 20;
    const int NL = in_sizes[0] / (P * 8);   // 20000
    const int NA = in_sizes[1] / (P * 8);   // 4000
    const int E_LL = in_sizes[2] / 2;
    const int E_AA = in_sizes[3] / 2;
    const int E_LA = in_sizes[4] / 2;

    const float* lane_pts = (const float*)d_in[0];
    const float* agent_hs = (const float*)d_in[1];
    const int*   e_ll = (const int*)d_in[2];
    const int*   e_aa = (const int*)d_in[3];
    const int*   e_la = (const int*)d_in[4];

    // weight/bias index table (in setup_inputs order)
    const int wIdx[12]  = {5, 7, 9, 11, 13, 15, 17, 19, 21, 23, 25, 27};
    const int Kreal[12] = {8, 128, 128, 8, 128, 128, 256, 128, 256, 128, 256, 128};
    const int Kc[12]    = {1, 4,   4,   1, 4,   4,   8,   4,   8,   4,   8,   4};
    const float* bptr[12];
    for (int i = 0; i < 12; ++i) bptr[i] = (const float*)d_in[wIdx[i] + 1];

    float* outLane  = (float*)d_out;
    float* outAgent = outLane + (size_t)NL * 128;

    // ---- workspace carve ----
    char* wp_cursor = (char*)d_ws;
    auto take = [&](size_t bytes) -> void* {
        void* r = (void*)wp_cursor;
        wp_cursor += (bytes + 255) & ~(size_t)255;
        return r;
    };
    __bf16* wpack[12];
    for (int i = 0; i < 12; ++i)
        wpack[i] = (__bf16*)take((size_t)Kc[i] * 8 * 32 * 16 * 2);

    const size_t maxRows = (size_t)NL * P;                       // 400000
    __bf16* xc     = (__bf16*)take(maxRows * 32 * 2);            // padded inputs
    __bf16* bufH   = (__bf16*)take(maxRows * 128 * 2);           // h1 / h2 (in-place)
    __bf16* hmax   = (__bf16*)take((size_t)NL * 128 * 2);        // pooled
    __bf16* laneB  = (__bf16*)take((size_t)NL * 128 * 2);        // lane feats bf16
    __bf16* agentB = (__bf16*)take((size_t)NA * 128 * 2);        // agent feats bf16
    float*  agg    = (float*)take((size_t)NL * 128 * 4);
    float*  deg    = (float*)take((size_t)NL * 4);
    __bf16* ccat   = (__bf16*)take((size_t)NL * 256 * 2);        // concat inputs
    __bf16* hg     = (__bf16*)take((size_t)NL * 128 * 2);        // GNN hidden

    // ---- pack all weights to bf16 WMMA B-fragment layout ----
    for (int i = 0; i < 12; ++i) {
        int total = Kc[i] * 8 * 32;
        vn_pack_weight<<<cdiv(total, 256), 256, 0, stream>>>(
            (const float*)d_in[wIdx[i]], wpack[i], Kreal[i], Kc[i]);
    }

    // ================= lane encoder =================
    vn_center_pad<<<cdiv((long)NL * P, 256), 256, 0, stream>>>(lane_pts, xc, NL, P);
    vn_gemm<1, true, false, true, false><<<cdiv(NL * P, 128), 256, 0, stream>>>(
        xc, 32, wpack[0], bptr[0], NL * P, nullptr, bufH, nullptr);
    vn_gemm<4, true, false, true, false><<<cdiv(NL * P, 128), 256, 0, stream>>>(
        bufH, 128, wpack[1], bptr[1], NL * P, nullptr, bufH, nullptr);   // in-place
    vn_maxpool<<<cdiv((long)NL * 128, 256), 256, 0, stream>>>(bufH, hmax, NL, P);
    vn_gemm<4, true, true, true, false><<<cdiv(NL, 128), 256, 0, stream>>>(
        hmax, 128, wpack[2], bptr[2], NL, outLane, laneB, nullptr);      // lane0

    // ================= agent encoder =================
    vn_center_pad<<<cdiv((long)NA * P, 256), 256, 0, stream>>>(agent_hs, xc, NA, P);
    vn_gemm<1, true, false, true, false><<<cdiv(NA * P, 128), 256, 0, stream>>>(
        xc, 32, wpack[3], bptr[3], NA * P, nullptr, bufH, nullptr);
    vn_gemm<4, true, false, true, false><<<cdiv(NA * P, 128), 256, 0, stream>>>(
        bufH, 128, wpack[4], bptr[4], NA * P, nullptr, bufH, nullptr);
    vn_maxpool<<<cdiv((long)NA * 128, 256), 256, 0, stream>>>(bufH, hmax, NA, P);
    vn_gemm<4, true, true, true, false><<<cdiv(NA, 128), 256, 0, stream>>>(
        hmax, 128, wpack[5], bptr[5], NA, outAgent, agentB, nullptr);    // agent0

    // ================= lane-lane GNN =================
    vn_zero_f32<<<cdiv((long)NL * 128, 256), 256, 0, stream>>>(agg, (long)NL * 128);
    vn_zero_f32<<<cdiv((long)NL, 256), 256, 0, stream>>>(deg, (long)NL);
    vn_scatter_add<<<cdiv((long)E_LL * 128, 256), 256, 0, stream>>>(
        e_ll, e_ll + E_LL, E_LL, laneB, agg, deg);
    vn_concat<<<cdiv((long)NL * 128, 256), 256, 0, stream>>>(laneB, agg, deg, ccat, NL);
    vn_gemm<8, false, false, true, false><<<cdiv(NL, 128), 256, 0, stream>>>(
        ccat, 256, wpack[6], bptr[6], NL, nullptr, hg, nullptr);
    vn_gemm<4, true, true, true, true><<<cdiv(NL, 128), 256, 0, stream>>>(
        hg, 128, wpack[7], bptr[7], NL, outLane, laneB, outLane);        // lane final

    // ================= agent-agent GNN =================
    vn_zero_f32<<<cdiv((long)NA * 128, 256), 256, 0, stream>>>(agg, (long)NA * 128);
    vn_zero_f32<<<cdiv((long)NA, 256), 256, 0, stream>>>(deg, (long)NA);
    vn_scatter_add<<<cdiv((long)E_AA * 128, 256), 256, 0, stream>>>(
        e_aa, e_aa + E_AA, E_AA, agentB, agg, deg);
    vn_concat<<<cdiv((long)NA * 128, 256), 256, 0, stream>>>(agentB, agg, deg, ccat, NA);
    vn_gemm<8, false, false, true, false><<<cdiv(NA, 128), 256, 0, stream>>>(
        ccat, 256, wpack[8], bptr[8], NA, nullptr, hg, nullptr);
    vn_gemm<4, true, true, true, true><<<cdiv(NA, 128), 256, 0, stream>>>(
        hg, 128, wpack[9], bptr[9], NA, outAgent, agentB, outAgent);     // agent1

    // ================= lane-agent GNN (source = updated lane) =================
    vn_zero_f32<<<cdiv((long)NA * 128, 256), 256, 0, stream>>>(agg, (long)NA * 128);
    vn_zero_f32<<<cdiv((long)NA, 256), 256, 0, stream>>>(deg, (long)NA);
    vn_scatter_add<<<cdiv((long)E_LA * 128, 256), 256, 0, stream>>>(
        e_la, e_la + E_LA, E_LA, laneB, agg, deg);
    vn_concat<<<cdiv((long)NA * 128, 256), 256, 0, stream>>>(agentB, agg, deg, ccat, NA);
    vn_gemm<8, false, false, true, false><<<cdiv(NA, 128), 256, 0, stream>>>(
        ccat, 256, wpack[10], bptr[10], NA, nullptr, hg, nullptr);
    vn_gemm<4, true, true, false, true><<<cdiv(NA, 128), 256, 0, stream>>>(
        hg, 128, wpack[11], bptr[11], NA, outAgent, nullptr, outAgent);  // agent final
}
